// Non_local_module_11184094839128
// MI455X (gfx1250) — compile-verified, matrix operand order checked by reading, not run on hardware
//
#include <hip/hip_runtime.h>
#include <hip/hip_bf16.h>

typedef __bf16 bf16;
typedef __attribute__((ext_vector_type(16))) __bf16 v16bf;
typedef __attribute__((ext_vector_type(8)))  float  v8f;
typedef int v4i __attribute__((vector_size(4 * sizeof(int))));

static constexpr int BB   = 4;     // batch
static constexpr int CIN  = 512;   // input channels
static constexpr int CK   = 256;   // q/k/v channels
static constexpr int OC   = 768;   // concatenated projection rows
static constexpr int NPOS = 4096;  // H*W

#if defined(__gfx1250__) && __has_builtin(__builtin_amdgcn_global_load_async_to_lds_b128)
#define USE_ASYNC_LDS 1
#else
#define USE_ASYNC_LDS 0
#endif

static __device__ __forceinline__ void wait_async_zero()
{
#if __has_builtin(__builtin_amdgcn_s_wait_asynccnt)
    __builtin_amdgcn_s_wait_asynccnt(0);
#elif defined(__gfx1250__)
    asm volatile("s_wait_asynccnt 0x0" ::: "memory");
#endif
}

// ---------------------------------------------------------------------------
// Generic bf16 WMMA GEMM:  C[M,N] = A[M,K] * B[K,N]  with B given as BT[N,K]
// (both operands K-contiguous).  128x128x32 macro tile, 8 waves, each wave
// computes a 32x64 patch via 2x4 v_wmma_f32_16x16x32_bf16 fragments.
// Double-buffered LDS; global->LDS via CDNA5 async loads when available.
// ---------------------------------------------------------------------------
template<bool OUT_F32, bool HAS_BIAS, bool HAS_RESID>
__global__ __launch_bounds__(256)
void gemm_bf16_wmma(const bf16* __restrict__ A,  long long aBatch,
                    const bf16* __restrict__ BT, long long btBatch,
                    void* __restrict__ Cout,     long long cBatch,
                    const float* __restrict__ bias,
                    const float* __restrict__ resid, long long rBatch,
                    int M, int N, int K)
{
    // 96-byte row pitch: every ds access stays 16B-aligned; banks well spread.
    __shared__ __align__(16) bf16 sA[2][128 * 48];
    __shared__ __align__(16) bf16 sB[2][128 * 48];

    const int bz = blockIdx.z;
    A  += (size_t)bz * aBatch;
    BT += (size_t)bz * btBatch;
    const float* residB = HAS_RESID ? (resid + (size_t)bz * rBatch) : nullptr;

    const int mblk = blockIdx.y * 128;
    const int nblk = blockIdx.x * 128;
    const int t    = threadIdx.x;
    const int lane = t & 31;
    const int wave = t >> 5;
    const int wrow = wave >> 1;   // 0..3 -> 32 output rows each
    const int wcol = wave & 1;    // 0..1 -> 64 output cols each
    const int lrow = lane & 15;
    const int g    = lane >> 4;   // half-wave group

    // Per-thread staging coordinates: 2 chunks of 8 halves per matrix per tile.
    int rowc[2], kcc[2];
    #pragma unroll
    for (int i = 0; i < 2; ++i) {
        int chunk = t * 2 + i;          // 512 chunks
        rowc[i] = chunk >> 2;           // 0..127
        kcc[i]  = (chunk & 3) << 3;     // 0,8,16,24
    }

    v8f acc[2][4];
    #pragma unroll
    for (int mi = 0; mi < 2; ++mi)
        #pragma unroll
        for (int ni = 0; ni < 4; ++ni)
            #pragma unroll
            for (int e = 0; e < 8; ++e) acc[mi][ni][e] = 0.0f;

    // Fragment loads per CDNA5 16-bit A/B layout:
    // lanes 0-15 hold K in {0..7,16..23}; lanes 16-31 hold K in {8..15,24..31}.
    auto compute_tile = [&](int buf) {
        v16bf af[2], bfr[4];
        #pragma unroll
        for (int mi = 0; mi < 2; ++mi) {
            int row = wrow * 32 + mi * 16 + lrow;
            const bf16* p = &sA[buf][row * 48 + g * 8];
            uint4* dst = (uint4*)&af[mi];
            dst[0] = *(const uint4*)(p);
            dst[1] = *(const uint4*)(p + 16);
        }
        #pragma unroll
        for (int ni = 0; ni < 4; ++ni) {
            int col = wcol * 64 + ni * 16 + lrow;
            const bf16* p = &sB[buf][col * 48 + g * 8];
            uint4* dst = (uint4*)&bfr[ni];
            dst[0] = *(const uint4*)(p);
            dst[1] = *(const uint4*)(p + 16);
        }
        #pragma unroll
        for (int mi = 0; mi < 2; ++mi)
            #pragma unroll
            for (int ni = 0; ni < 4; ++ni)
                acc[mi][ni] = __builtin_amdgcn_wmma_f32_16x16x32_bf16(
                    false, af[mi], false, bfr[ni], (short)0, acc[mi][ni],
                    false, false);
    };

#if USE_ASYNC_LDS
    // Issue a 128x32 A tile + 128x32 BT tile as async global->LDS copies.
    auto issue_tile = [&](int buf, int kt) {
        #pragma unroll
        for (int i = 0; i < 2; ++i) {
            const bf16* ga = A  + (size_t)(mblk + rowc[i]) * K + kt + kcc[i];
            const bf16* gb = BT + (size_t)(nblk + rowc[i]) * K + kt + kcc[i];
            __builtin_amdgcn_global_load_async_to_lds_b128(
                (v4i*)ga, (v4i*)&sA[buf][rowc[i] * 48 + kcc[i]], 0, 0);
            __builtin_amdgcn_global_load_async_to_lds_b128(
                (v4i*)gb, (v4i*)&sB[buf][rowc[i] * 48 + kcc[i]], 0, 0);
        }
    };

    issue_tile(0, 0);
    wait_async_zero();
    __syncthreads();
    int cur = 0;
    for (int kt = 0; kt < K; kt += 32) {
        if (kt + 32 < K) issue_tile(cur ^ 1, kt + 32);  // overlap with WMMA
        compute_tile(cur);
        wait_async_zero();       // next tile landed in the other buffer
        __syncthreads();
        cur ^= 1;
    }
#else
    // Fallback: register-staged double buffering.
    uint4 ra[2], rb[2];
    auto load_regs = [&](int kt) {
        #pragma unroll
        for (int i = 0; i < 2; ++i) {
            ra[i] = *(const uint4*)(A  + (size_t)(mblk + rowc[i]) * K + kt + kcc[i]);
            rb[i] = *(const uint4*)(BT + (size_t)(nblk + rowc[i]) * K + kt + kcc[i]);
        }
    };
    auto store_lds = [&](int buf) {
        #pragma unroll
        for (int i = 0; i < 2; ++i) {
            *(uint4*)&sA[buf][rowc[i] * 48 + kcc[i]] = ra[i];
            *(uint4*)&sB[buf][rowc[i] * 48 + kcc[i]] = rb[i];
        }
    };
    load_regs(0);
    store_lds(0);
    __syncthreads();
    int cur = 0;
    for (int kt = 0; kt < K; kt += 32) {
        if (kt + 32 < K) {
            load_regs(kt + 32);                        // overlap with WMMA below
            __builtin_prefetch(A + (size_t)(mblk + (t >> 1)) * K + kt + 64, 0, 3);
        }
        compute_tile(cur);
        if (kt + 32 < K) store_lds(cur ^ 1);
        __syncthreads();
        cur ^= 1;
    }
#endif

    // Epilogue.  C/D layout: VGPR r holds row (8*g + r), lane%16 gives column.
    #pragma unroll
    for (int mi = 0; mi < 2; ++mi) {
        #pragma unroll
        for (int ni = 0; ni < 4; ++ni) {
            int colG = nblk + wcol * 64 + ni * 16 + lrow;
            #pragma unroll
            for (int r = 0; r < 8; ++r) {
                int rowG = mblk + wrow * 32 + mi * 16 + g * 8 + r;
                float v = acc[mi][ni][r];
                if (HAS_BIAS)  v += bias[rowG];
                size_t idx = (size_t)rowG * N + colG;
                if (HAS_RESID) v += residB[idx];
                if (OUT_F32) ((float*)Cout)[(size_t)bz * cBatch + idx] = v;
                else         ((bf16*) Cout)[(size_t)bz * cBatch + idx] = (bf16)v;
            }
        }
    }
}

// ---------------------------------------------------------------------------
// Parameter prep: concat weights/bias to bf16 GEMM layouts.
// ---------------------------------------------------------------------------
__global__ __launch_bounds__(256)
void prep_params(const float* __restrict__ wq, const float* __restrict__ wk,
                 const float* __restrict__ wv, const float* __restrict__ wW,
                 const float* __restrict__ bq, const float* __restrict__ bk,
                 const float* __restrict__ bv,
                 bf16* __restrict__ wcat, bf16* __restrict__ wWb,
                 float* __restrict__ bcat)
{
    int i = blockIdx.x * blockDim.x + threadIdx.x;
    if (i < CK * CIN)            wcat[i] = (bf16)wq[i];
    else if (i < 2 * CK * CIN)   wcat[i] = (bf16)wk[i - CK * CIN];
    else if (i < OC * CIN)       wcat[i] = (bf16)wv[i - 2 * CK * CIN];
    if (i < CIN * CK)            wWb[i]  = (bf16)wW[i];
    if (i < CK)                  bcat[i] = bq[i];
    else if (i < 2 * CK)         bcat[i] = bk[i - CK];
    else if (i < OC)             bcat[i] = bv[i - 2 * CK];
}

// x [B,C,N] f32 -> xT [B,N,C] bf16 (K-contiguous operand for the projections)
__global__ __launch_bounds__(256)
void prep_x(const float* __restrict__ x, bf16* __restrict__ xT)
{
    size_t i = (size_t)blockIdx.x * blockDim.x + threadIdx.x;
    if (i >= (size_t)BB * CIN * NPOS) return;
    int n = (int)(i % NPOS);
    int c = (int)((i / NPOS) % CIN);
    int b = (int)(i / ((size_t)CIN * NPOS));
    xT[((size_t)b * NPOS + n) * CIN + c] = (bf16)x[i];
}

// Per-channel BN stats over (B, N) for the first 512 rows of z; fused scale/shift.
__global__ __launch_bounds__(256)
void bn_stats(const bf16* __restrict__ z,
              const float* __restrict__ gq, const float* __restrict__ betaq,
              const float* __restrict__ gk, const float* __restrict__ betak,
              float* __restrict__ scale, float* __restrict__ shift)
{
    __shared__ float rs[256], rs2[256];
    int ch = blockIdx.x;                 // 0..511
    float s = 0.f, s2 = 0.f;
    for (int b = 0; b < BB; ++b) {
        const bf16* p = z + (size_t)b * OC * NPOS + (size_t)ch * NPOS;
        for (int n = threadIdx.x; n < NPOS; n += 256) {
            float v = (float)p[n];
            s += v; s2 += v * v;
        }
    }
    rs[threadIdx.x] = s; rs2[threadIdx.x] = s2;
    __syncthreads();
    for (int off = 128; off > 0; off >>= 1) {
        if (threadIdx.x < off) {
            rs[threadIdx.x]  += rs[threadIdx.x + off];
            rs2[threadIdx.x] += rs2[threadIdx.x + off];
        }
        __syncthreads();
    }
    if (threadIdx.x == 0) {
        float cnt = (float)BB * NPOS;
        float m   = rs[0] / cnt;
        float var = rs2[0] / cnt - m * m;        // biased var, like jnp.var
        float gamma = (ch < CK) ? gq[ch] : gk[ch - CK];
        float beta  = (ch < CK) ? betaq[ch] : betak[ch - CK];
        float sc = gamma * rsqrtf(var + 1e-5f);
        scale[ch] = sc;
        shift[ch] = beta - m * sc;
    }
}

// BN + ReLU, writing qT/kT [B,N,CK] bf16 (K-contiguous for the attention GEMM).
__global__ __launch_bounds__(256)
void bn_apply(const bf16* __restrict__ z,
              const float* __restrict__ scale, const float* __restrict__ shift,
              bf16* __restrict__ qT, bf16* __restrict__ kT)
{
    size_t i = (size_t)blockIdx.x * blockDim.x + threadIdx.x;
    if (i >= (size_t)BB * (2 * CK) * NPOS) return;
    int n  = (int)(i % NPOS);
    int ch = (int)((i / NPOS) % (2 * CK));
    int b  = (int)(i / ((size_t)(2 * CK) * NPOS));
    float v = (float)z[(size_t)b * OC * NPOS + (size_t)ch * NPOS + n];
    v = v * scale[ch] + shift[ch];
    v = fmaxf(v, 0.f);
    if (ch < CK) qT[((size_t)b * NPOS + n) * CK + ch]        = (bf16)v;
    else         kT[((size_t)b * NPOS + n) * CK + (ch - CK)] = (bf16)v;
}

// In-place row softmax over bf16 logits; 1/sqrt(Ck) folded into the logits.
__global__ __launch_bounds__(256)
void softmax_rows(bf16* __restrict__ att, float lscale)
{
    __shared__ float red[256];
    bf16* p = att + (size_t)blockIdx.x * NPOS;
    float v[16];
    #pragma unroll
    for (int i = 0; i < 16; ++i)
        v[i] = (float)p[threadIdx.x + i * 256] * lscale;

    float mx = v[0];
    #pragma unroll
    for (int i = 1; i < 16; ++i) mx = fmaxf(mx, v[i]);
    red[threadIdx.x] = mx;
    __syncthreads();
    for (int off = 128; off > 0; off >>= 1) {
        if (threadIdx.x < off)
            red[threadIdx.x] = fmaxf(red[threadIdx.x], red[threadIdx.x + off]);
        __syncthreads();
    }
    mx = red[0];
    __syncthreads();

    float s = 0.f;
    #pragma unroll
    for (int i = 0; i < 16; ++i) { v[i] = __expf(v[i] - mx); s += v[i]; }
    red[threadIdx.x] = s;
    __syncthreads();
    for (int off = 128; off > 0; off >>= 1) {
        if (threadIdx.x < off) red[threadIdx.x] += red[threadIdx.x + off];
        __syncthreads();
    }
    float inv = 1.0f / red[0];
    #pragma unroll
    for (int i = 0; i < 16; ++i)
        p[threadIdx.x + i * 256] = (bf16)(v[i] * inv);
}

// ---------------------------------------------------------------------------
extern "C" void kernel_launch(void* const* d_in, const int* in_sizes, int n_in,
                              void* d_out, int out_size, void* d_ws, size_t ws_size,
                              hipStream_t stream)
{
    const float* x     = (const float*)d_in[0];
    const float* wq    = (const float*)d_in[1];
    const float* bq    = (const float*)d_in[2];
    const float* gq    = (const float*)d_in[3];
    const float* betaq = (const float*)d_in[4];
    const float* wk    = (const float*)d_in[5];
    const float* bk    = (const float*)d_in[6];
    const float* gk    = (const float*)d_in[7];
    const float* betak = (const float*)d_in[8];
    const float* wv    = (const float*)d_in[9];
    const float* bv    = (const float*)d_in[10];
    const float* wW    = (const float*)d_in[11];
    const float* bW    = (const float*)d_in[12];
    float* out = (float*)d_out;

    char* ws = (char*)d_ws;
    size_t off = 0;
    auto alloc = [&](size_t bytes) -> char* {
        char* p = ws + off;
        off = (off + bytes + 255) & ~(size_t)255;
        return p;
    };
    bf16*  xT    = (bf16*)alloc((size_t)BB * NPOS * CIN * 2);  // [B,N,C]
    bf16*  z     = (bf16*)alloc((size_t)BB * OC * NPOS * 2);   // [B,768,N] (rows 512.. = v)
    bf16*  qT    = (bf16*)alloc((size_t)BB * NPOS * CK * 2);   // [B,N,256]
    bf16*  kT    = (bf16*)alloc((size_t)BB * NPOS * CK * 2);   // [B,N,256]
    bf16*  att   = (bf16*)alloc((size_t)NPOS * NPOS * 2);      // [N,N], reused per batch
    bf16*  ctxT  = (bf16*)alloc((size_t)BB * NPOS * CK * 2);   // [B,N,256]
    bf16*  wcat  = (bf16*)alloc((size_t)OC * CIN * 2);         // [768,512]
    bf16*  wWb   = (bf16*)alloc((size_t)CIN * CK * 2);         // [512,256]
    float* bcat  = (float*)alloc(OC * 4);
    float* scale = (float*)alloc(512 * 4);
    float* shift = (float*)alloc(512 * 4);

    // 1) parameter + input prep
    prep_params<<<(OC * CIN + 255) / 256, 256, 0, stream>>>(
        wq, wk, wv, wW, bq, bk, bv, wcat, wWb, bcat);
    prep_x<<<(int)(((size_t)BB * CIN * NPOS + 255) / 256), 256, 0, stream>>>(x, xT);

    // 2) fused projections: z[b] = wcat * x[b]  (M=768, N=4096, K=512) + bias
    gemm_bf16_wmma<false, true, false><<<dim3(NPOS / 128, OC / 128, BB), 256, 0, stream>>>(
        wcat, 0, xT, (long long)NPOS * CIN, z, (long long)OC * NPOS,
        bcat, nullptr, 0, OC, NPOS, CIN);

    // 3) BN stats + apply (q,k -> ReLU(BN)), transposed bf16 outputs
    bn_stats<<<512, 256, 0, stream>>>(z, gq, betaq, gk, betak, scale, shift);
    bn_apply<<<(int)(((size_t)BB * 2 * CK * NPOS + 255) / 256), 256, 0, stream>>>(
        z, scale, shift, qT, kT);

    const float lscale = 0.0625f;  // 256^-0.5
    for (int b = 0; b < BB; ++b) {
        const bf16* kTb = kT + (size_t)b * NPOS * CK;
        const bf16* qTb = qT + (size_t)b * NPOS * CK;
        const bf16* vb  = z  + (size_t)b * OC * NPOS + (size_t)(2 * CK) * NPOS; // [256,N]
        bf16* ctxb      = ctxT + (size_t)b * NPOS * CK;

        // att[n,m] = sum_c kT[n,c] * qT[m,c]   (M=N=4096, K=256)
        gemm_bf16_wmma<false, false, false><<<dim3(NPOS / 128, NPOS / 128, 1), 256, 0, stream>>>(
            kTb, 0, qTb, 0, att, 0, nullptr, nullptr, 0, NPOS, NPOS, CK);

        softmax_rows<<<NPOS, 256, 0, stream>>>(att, lscale);

        // ctxT[n,c] = sum_m att[n,m] * v[c,m]  (M=4096, N=256, K=4096; BT = v c-major)
        gemm_bf16_wmma<false, false, false><<<dim3(CK / 128, NPOS / 128, 1), 256, 0, stream>>>(
            att, 0, vb, 0, ctxb, 0, nullptr, nullptr, 0, NPOS, CK, NPOS);
    }

    // 4) out[b,co,n] = sum_cv wW[co,cv]*ctx[b,n,cv] + bW[co] + x[b,co,n]
    gemm_bf16_wmma<true, true, true><<<dim3(NPOS / 128, CIN / 128, BB), 256, 0, stream>>>(
        wWb, 0, ctxT, (long long)NPOS * CK, out, (long long)CIN * NPOS,
        bW, x, (long long)CIN * NPOS, CIN, NPOS, CK);
}
